// BaselineMiniTransformer_88441966559315
// MI455X (gfx1250) — matrix operand order repaired and checked by reading, not compiled, not run on hardware
//
#include <hip/hip_runtime.h>
#include <hip/hip_bf16.h>
#include <math.h>

// ---------------------------------------------------------------------------
// MI455X (gfx1250) mini-transformer block, bf16 WMMA path, f32 accumulate.
// Round 3: enforce the software pipeline with __builtin_amdgcn_sched_barrier
// between the next-step fragment loads and the current-step WMMAs, so the
// scheduler cannot collapse the double buffer (which previously produced
// s_wait_loadcnt 0x0 before every v_wmma).
// ---------------------------------------------------------------------------

typedef __bf16 bf16;
typedef __attribute__((ext_vector_type(16))) __bf16 v16bf;
typedef __attribute__((ext_vector_type(8)))  __bf16 v8bf;
typedef __attribute__((ext_vector_type(8)))  float  v8f;

#if defined(__has_builtin)
#if __has_builtin(__builtin_amdgcn_sched_barrier)
#define SCHED_FENCE() __builtin_amdgcn_sched_barrier(0)
#endif
#endif
#ifndef SCHED_FENCE
#define SCHED_FENCE()
#endif

#define D_MODEL 1024
#define D_FF    4096
#define N_HEADS 16
#define D_HEAD  64
#define BSZ     2
#define TLEN    2048
#define ROWS    (BSZ * TLEN)   // 4096 token rows

__device__ __forceinline__ v8f v8f_zero() {
    v8f z;
#pragma unroll
    for (int i = 0; i < 8; ++i) z[i] = 0.0f;
    return z;
}

// A-matrix fragment (16x32 bf16), row-major source with leading dim `ld`.
// Lanes 0-15 hold row m, K={0..15}; lanes 16-31 hold row m,
// K={16..31} (matches load_b_frag).
__device__ __forceinline__ v16bf load_a_frag(const bf16* __restrict__ base,
                                             int ld, int row0, int k0) {
    const int lane = threadIdx.x & 31;
    const bf16* p = base + (size_t)(row0 + (lane & 15)) * ld + k0 + ((lane & 16) ? 16 : 0);
    v8bf lo = *(const v8bf*)(p);        // K j..j+7
    v8bf hi = *(const v8bf*)(p + 8);    // K j+8..j+15
    v16bf r;
#pragma unroll
    for (int i = 0; i < 8; ++i) { r[i] = lo[i]; r[i + 8] = hi[i]; }
    return r;
}

// Same as load_a_frag but from an f32 source (LDS softmax probabilities).
__device__ __forceinline__ v16bf load_a_frag_f32(const float* __restrict__ base,
                                                 int ld, int row0, int k0) {
    const int lane = threadIdx.x & 31;
    const float* p = base + (size_t)(row0 + (lane & 15)) * ld + k0 + ((lane & 16) ? 16 : 0);
    v16bf r;
#pragma unroll
    for (int i = 0; i < 8; ++i) { r[i] = (bf16)p[i]; r[i + 8] = (bf16)p[i + 8]; }
    return r;
}

// B-matrix fragment (32x16 bf16) sourced from a row-major [N,K] matrix
// (i.e. computing X @ W^T). Lane n holds W[n][k0 + 0..15] (lanes 0-15) or
// W[n][k0 + 16..31] (lanes 16-31): one contiguous 32-byte load.
__device__ __forceinline__ v16bf load_b_frag(const bf16* __restrict__ base,
                                             int ld, int col0, int k0) {
    const int lane = threadIdx.x & 31;
    const bf16* p = base + (size_t)(col0 + (lane & 15)) * ld + k0 + ((lane & 16) ? 16 : 0);
    return *(const v16bf*)(p);
}

__device__ __forceinline__ v8f wmma_bf16(v16bf a, v16bf b, v8f c) {
    return __builtin_amdgcn_wmma_f32_16x16x32_bf16(false, a, false, b,
                                                   (short)0, c, false, false);
}

// One wave computes a 16(M) x 64(N) tile. Software pipelined with an explicit
// scheduling fence: loads for K-step k+32 are pinned before the WMMAs of step
// k, so waitcnt insertion only drains the previous step's loads.
template <int K>
__device__ __forceinline__ void gemm_tile_16x64(const bf16* __restrict__ A,
                                                const bf16* __restrict__ W,
                                                int m0, int n0, v8f acc[4]) {
#pragma unroll
    for (int j = 0; j < 4; ++j) acc[j] = v8f_zero();
    v16bf a_c  = load_a_frag(A, K, m0, 0);
    v16bf b_c0 = load_b_frag(W, K, n0 +  0, 0);
    v16bf b_c1 = load_b_frag(W, K, n0 + 16, 0);
    v16bf b_c2 = load_b_frag(W, K, n0 + 32, 0);
    v16bf b_c3 = load_b_frag(W, K, n0 + 48, 0);
#pragma unroll 2
    for (int k0 = 32; k0 < K; k0 += 32) {
        v16bf a_n  = load_a_frag(A, K, m0, k0);
        v16bf b_n0 = load_b_frag(W, K, n0 +  0, k0);
        v16bf b_n1 = load_b_frag(W, K, n0 + 16, k0);
        v16bf b_n2 = load_b_frag(W, K, n0 + 32, k0);
        v16bf b_n3 = load_b_frag(W, K, n0 + 48, k0);
        // speculative prefetch of A three K-steps ahead of the loads (faults are dropped)
        __builtin_prefetch(A + (size_t)(m0 + (threadIdx.x & 15)) * K + k0 + 96, 0, 1);
        SCHED_FENCE();   // keep next-step loads above current-step WMMAs
        acc[0] = wmma_bf16(a_c, b_c0, acc[0]);
        acc[1] = wmma_bf16(a_c, b_c1, acc[1]);
        acc[2] = wmma_bf16(a_c, b_c2, acc[2]);
        acc[3] = wmma_bf16(a_c, b_c3, acc[3]);
        SCHED_FENCE();   // keep WMMAs above the next iteration's loads
        a_c = a_n; b_c0 = b_n0; b_c1 = b_n1; b_c2 = b_n2; b_c3 = b_n3;
    }
    acc[0] = wmma_bf16(a_c, b_c0, acc[0]);
    acc[1] = wmma_bf16(a_c, b_c1, acc[1]);
    acc[2] = wmma_bf16(a_c, b_c2, acc[2]);
    acc[3] = wmma_bf16(a_c, b_c3, acc[3]);
}

// ------------------------------- kernels -----------------------------------

__global__ void k_cvt_bf16(const float* __restrict__ in, bf16* __restrict__ out, int n4) {
    int i = blockIdx.x * blockDim.x + threadIdx.x;
    if (i < n4) {
        float4 v = ((const float4*)in)[i];
        out[4 * i + 0] = (bf16)v.x;
        out[4 * i + 1] = (bf16)v.y;
        out[4 * i + 2] = (bf16)v.z;
        out[4 * i + 3] = (bf16)v.w;
    }
}

// LayerNorm over 1024 cols; one 256-thread block per row; bf16 output.
__global__ void k_layernorm(const float* __restrict__ x, const float* __restrict__ g,
                            const float* __restrict__ bta, bf16* __restrict__ out) {
    const int row = blockIdx.x;
    const int tid = threadIdx.x;
    const float* xr = x + (size_t)row * D_MODEL;
    float v[4];
    float s = 0.0f;
#pragma unroll
    for (int i = 0; i < 4; ++i) { v[i] = xr[tid + 256 * i]; s += v[i]; }
    __shared__ float red[8];
#pragma unroll
    for (int off = 16; off > 0; off >>= 1) s += __shfl_xor(s, off, 32);
    if ((tid & 31) == 0) red[tid >> 5] = s;
    __syncthreads();
    float mean = 0.0f;
#pragma unroll
    for (int w = 0; w < 8; ++w) mean += red[w];
    mean *= (1.0f / (float)D_MODEL);
    __syncthreads();
    float s2 = 0.0f;
#pragma unroll
    for (int i = 0; i < 4; ++i) { float d = v[i] - mean; s2 += d * d; }
#pragma unroll
    for (int off = 16; off > 0; off >>= 1) s2 += __shfl_xor(s2, off, 32);
    if ((tid & 31) == 0) red[tid >> 5] = s2;
    __syncthreads();
    float var = 0.0f;
#pragma unroll
    for (int w = 0; w < 8; ++w) var += red[w];
    var *= (1.0f / (float)D_MODEL);
    const float rstd = rsqrtf(var + 1e-5f);
    bf16* orow = out + (size_t)row * D_MODEL;
#pragma unroll
    for (int i = 0; i < 4; ++i) {
        int c = tid + 256 * i;
        orow[c] = (bf16)((v[i] - mean) * rstd * g[c] + bta[c]);
    }
}

// Q/K/V projection: y = h @ W^T, 4096x1024x1024. v_mode=0 -> [B,H,T,Dh];
// v_mode=1 -> transposed [B,H,Dh,T] (for the P@V B-operand).
__global__ void k_gemm_qkv(const bf16* __restrict__ h, const bf16* __restrict__ w,
                           bf16* __restrict__ out, int v_mode) {
    const int wid = blockIdx.x * 8 + (threadIdx.x >> 5);
    const int m0 = (wid >> 4) * 16;
    const int n0 = (wid & 15) * 64;
    v8f acc[4];
    gemm_tile_16x64<D_MODEL>(h, w, m0, n0, acc);
    const int lane = threadIdx.x & 31;
    const int head = n0 / D_HEAD;            // 64-wide tile == one head
    const int rbase = m0 + ((lane & 16) ? 8 : 0);
#pragma unroll
    for (int j = 0; j < 4; ++j) {
        int dh = 16 * j + (lane & 15);
#pragma unroll
        for (int r = 0; r < 8; ++r) {
            int m = rbase + r;
            int b = m >> 11, t = m & (TLEN - 1);
            size_t idx = v_mode
                ? ((size_t)((b * N_HEADS + head) * D_HEAD + dh)) * TLEN + t
                : ((size_t)((b * N_HEADS + head) * TLEN + t)) * D_HEAD + dh;
            out[idx] = (bf16)acc[j][r];
        }
    }
}

// Attention: one 128-thread (4-wave) block per (b, h, 16-query tile).
// LDS: 16 x 2048 f32 scores (128 KB), softmaxed in place, cvt on A-load.
__global__ void k_attention(const bf16* __restrict__ q, const bf16* __restrict__ k,
                            const bf16* __restrict__ vT, bf16* __restrict__ attn) {
    extern __shared__ float S[];             // [16][2048]
    const int qt = blockIdx.x & 127;
    const int bh = blockIdx.x >> 7;
    const int b = bh >> 4, h = bh & 15;
    const bf16* Qh = q + ((size_t)bh * TLEN + qt * 16) * D_HEAD;
    const bf16* Kh = k + (size_t)bh * TLEN * D_HEAD;
    const bf16* Vt = vT + (size_t)bh * D_HEAD * TLEN;
    const int tid = threadIdx.x;
    const int w = tid >> 5, lane = tid & 31;
    const int rb = (lane & 16) ? 8 : 0;

    // Phase 1: scores S = Q K^T / sqrt(64). Wave w owns columns [512w, 512w+512).
    // Pipelined over the 32 column tiles.
    v16bf qa0 = load_a_frag(Qh, D_HEAD, 0, 0);
    v16bf qa1 = load_a_frag(Qh, D_HEAD, 0, 32);
    v16bf kb0 = load_b_frag(Kh, D_HEAD, w * 512, 0);
    v16bf kb1 = load_b_frag(Kh, D_HEAD, w * 512, 32);
#pragma unroll 2
    for (int nt = 0; nt < 31; ++nt) {
        int n0 = w * 512 + nt * 16;
        v16bf nb0 = load_b_frag(Kh, D_HEAD, n0 + 16, 0);
        v16bf nb1 = load_b_frag(Kh, D_HEAD, n0 + 16, 32);
        SCHED_FENCE();
        v8f acc = v8f_zero();
        acc = wmma_bf16(qa0, kb0, acc);
        acc = wmma_bf16(qa1, kb1, acc);
        int col = n0 + (lane & 15);
#pragma unroll
        for (int r = 0; r < 8; ++r) S[(rb + r) * TLEN + col] = acc[r] * 0.125f;
        SCHED_FENCE();
        kb0 = nb0; kb1 = nb1;
    }
    {
        v8f acc = v8f_zero();
        acc = wmma_bf16(qa0, kb0, acc);
        acc = wmma_bf16(qa1, kb1, acc);
        int col = w * 512 + 31 * 16 + (lane & 15);
#pragma unroll
        for (int r = 0; r < 8; ++r) S[(rb + r) * TLEN + col] = acc[r] * 0.125f;
    }
    __syncthreads();

    // Phase 2: row softmax, in place. Wave w owns rows 4w..4w+3.
    for (int rr = 0; rr < 4; ++rr) {
        float* Sr = S + (w * 4 + rr) * TLEN;
        float m = -3.0e38f;
        for (int i = lane; i < TLEN; i += 32) m = fmaxf(m, Sr[i]);
#pragma unroll
        for (int off = 16; off > 0; off >>= 1) m = fmaxf(m, __shfl_xor(m, off, 32));
        float sum = 0.0f;
        for (int i = lane; i < TLEN; i += 32) sum += __expf(Sr[i] - m);
#pragma unroll
        for (int off = 16; off > 0; off >>= 1) sum += __shfl_xor(sum, off, 32);
        float inv = 1.0f / sum;
        for (int i = lane; i < TLEN; i += 32) Sr[i] = __expf(Sr[i] - m) * inv;
    }
    __syncthreads();

    // Phase 3: out = P @ V. Wave w owns dh tile [16w, 16w+16). Pipelined on V.
    v8f acc = v8f_zero();
    v16bf vb = load_b_frag(Vt, TLEN, w * 16, 0);
#pragma unroll 2
    for (int t0 = 0; t0 < TLEN - 32; t0 += 32) {
        v16bf vn = load_b_frag(Vt, TLEN, w * 16, t0 + 32);
        v16bf a = load_a_frag_f32(S, TLEN, 0, t0);      // ds_load + cvt
        SCHED_FENCE();
        acc = wmma_bf16(a, vb, acc);
        SCHED_FENCE();
        vb = vn;
    }
    {
        v16bf a = load_a_frag_f32(S, TLEN, 0, TLEN - 32);
        acc = wmma_bf16(a, vb, acc);
    }
    const int c = h * D_HEAD + w * 16 + (lane & 15);
#pragma unroll
    for (int r = 0; r < 8; ++r) {
        int m = b * TLEN + qt * 16 + rb + r;
        attn[(size_t)m * D_MODEL + c] = (bf16)acc[r];
    }
}

// x1 = attn @ wo^T + x  (f32 residual stream)
__global__ void k_gemm_oproj(const bf16* __restrict__ a, const bf16* __restrict__ wo,
                             const float* __restrict__ x, float* __restrict__ x1) {
    const int wid = blockIdx.x * 8 + (threadIdx.x >> 5);
    const int m0 = (wid >> 4) * 16;
    const int n0 = (wid & 15) * 64;
    v8f acc[4];
    gemm_tile_16x64<D_MODEL>(a, wo, m0, n0, acc);
    const int lane = threadIdx.x & 31;
    const int rbase = m0 + ((lane & 16) ? 8 : 0);
#pragma unroll
    for (int j = 0; j < 4; ++j) {
        int c = n0 + 16 * j + (lane & 15);
#pragma unroll
        for (int r = 0; r < 8; ++r) {
            size_t idx = (size_t)(rbase + r) * D_MODEL + c;
            x1[idx] = acc[j][r] + x[idx];
        }
    }
}

// ff1 = gelu(h2 @ w1^T + b1)  -> bf16 [4096, 4096]
__global__ void k_gemm_ff1(const bf16* __restrict__ h2, const bf16* __restrict__ w1,
                           const float* __restrict__ b1, bf16* __restrict__ out) {
    const int wid = blockIdx.x * 8 + (threadIdx.x >> 5);
    const int m0 = (wid >> 6) * 16;
    const int n0 = (wid & 63) * 64;
    v8f acc[4];
    gemm_tile_16x64<D_MODEL>(h2, w1, m0, n0, acc);
    const int lane = threadIdx.x & 31;
    const int rbase = m0 + ((lane & 16) ? 8 : 0);
#pragma unroll
    for (int j = 0; j < 4; ++j) {
        int c = n0 + 16 * j + (lane & 15);
        float bias = b1[c];
#pragma unroll
        for (int r = 0; r < 8; ++r) {
            float v = acc[j][r] + bias;
            float gl = 0.5f * v * (1.0f + erff(v * 0.70710678118654752f));  // exact GELU
            out[(size_t)(rbase + r) * D_FF + c] = (bf16)gl;
        }
    }
}

// y = ff1 @ w2^T + b2 + x1  -> f32 output
__global__ void k_gemm_ff2(const bf16* __restrict__ f1, const bf16* __restrict__ w2,
                           const float* __restrict__ b2, const float* __restrict__ x1,
                           float* __restrict__ y) {
    const int wid = blockIdx.x * 8 + (threadIdx.x >> 5);
    const int m0 = (wid >> 4) * 16;
    const int n0 = (wid & 15) * 64;
    v8f acc[4];
    gemm_tile_16x64<D_FF>(f1, w2, m0, n0, acc);
    const int lane = threadIdx.x & 31;
    const int rbase = m0 + ((lane & 16) ? 8 : 0);
#pragma unroll
    for (int j = 0; j < 4; ++j) {
        int c = n0 + 16 * j + (lane & 15);
        float bias = b2[c];
#pragma unroll
        for (int r = 0; r < 8; ++r) {
            size_t idx = (size_t)(rbase + r) * D_MODEL + c;
            y[idx] = acc[j][r] + bias + x1[idx];
        }
    }
}

// ------------------------------- launcher ----------------------------------

extern "C" void kernel_launch(void* const* d_in, const int* in_sizes, int n_in,
                              void* d_out, int out_size, void* d_ws, size_t ws_size,
                              hipStream_t stream) {
    const float* x     = (const float*)d_in[0];
    const float* wq    = (const float*)d_in[1];
    const float* wk    = (const float*)d_in[2];
    const float* wv    = (const float*)d_in[3];
    const float* wo    = (const float*)d_in[4];
    const float* w1    = (const float*)d_in[5];
    const float* b1    = (const float*)d_in[6];
    const float* w2    = (const float*)d_in[7];
    const float* b2    = (const float*)d_in[8];
    const float* ln1_g = (const float*)d_in[9];
    const float* ln1_b = (const float*)d_in[10];
    const float* ln2_g = (const float*)d_in[11];
    const float* ln2_b = (const float*)d_in[12];
    float* out = (float*)d_out;

    const size_t MB = 1024ull * 1024ull;
    char* ws = (char*)d_ws;
    bf16* wq_b   = (bf16*)(ws + 0 * MB);     // 1M bf16 = 2 MB
    bf16* wk_b   = (bf16*)(ws + 2 * MB);
    bf16* wv_b   = (bf16*)(ws + 4 * MB);
    bf16* wo_b   = (bf16*)(ws + 6 * MB);
    bf16* w1_b   = (bf16*)(ws + 8 * MB);     // 4M bf16 = 8 MB
    bf16* w2_b   = (bf16*)(ws + 16 * MB);    // 8 MB
    bf16* h_b    = (bf16*)(ws + 24 * MB);    // ln1 out, 8 MB
    bf16* q_b    = (bf16*)(ws + 32 * MB);    // [B,H,T,Dh], 8 MB
    bf16* k_b    = (bf16*)(ws + 40 * MB);
    bf16* vT_b   = (bf16*)(ws + 48 * MB);    // [B,H,Dh,T]
    bf16* attn_b = (bf16*)(ws + 56 * MB);    // [B*T, C]
    float* x1    = (float*)(ws + 64 * MB);   // residual stream, 16 MB
    bf16* h2_b   = (bf16*)(ws + 80 * MB);    // ln2 out, 8 MB
    bf16* ff1_b  = (bf16*)(ws + 88 * MB);    // [4096,4096], 32 MB -> ends at 120 MB

    // 1) weights -> bf16
    k_cvt_bf16<<<1024, 256, 0, stream>>>(wq, wq_b, (D_MODEL * D_MODEL) / 4);
    k_cvt_bf16<<<1024, 256, 0, stream>>>(wk, wk_b, (D_MODEL * D_MODEL) / 4);
    k_cvt_bf16<<<1024, 256, 0, stream>>>(wv, wv_b, (D_MODEL * D_MODEL) / 4);
    k_cvt_bf16<<<1024, 256, 0, stream>>>(wo, wo_b, (D_MODEL * D_MODEL) / 4);
    k_cvt_bf16<<<4096, 256, 0, stream>>>(w1, w1_b, (D_FF * D_MODEL) / 4);
    k_cvt_bf16<<<4096, 256, 0, stream>>>(w2, w2_b, (D_MODEL * D_FF) / 4);

    // 2) LN1
    k_layernorm<<<ROWS, 256, 0, stream>>>(x, ln1_g, ln1_b, h_b);

    // 3) Q,K,V projections (4096 waves each -> 512 blocks of 8 waves)
    k_gemm_qkv<<<512, 256, 0, stream>>>(h_b, wq_b, q_b, 0);
    k_gemm_qkv<<<512, 256, 0, stream>>>(h_b, wk_b, k_b, 0);
    k_gemm_qkv<<<512, 256, 0, stream>>>(h_b, wv_b, vT_b, 1);

    // 4) attention: B*H*(T/16) = 4096 blocks, 128 KB dynamic LDS
    k_attention<<<BSZ * N_HEADS * (TLEN / 16), 128, 16 * TLEN * sizeof(float), stream>>>(
        q_b, k_b, vT_b, attn_b);

    // 5) output projection + residual
    k_gemm_oproj<<<512, 256, 0, stream>>>(attn_b, wo_b, x, x1);

    // 6) LN2
    k_layernorm<<<ROWS, 256, 0, stream>>>(x1, ln2_g, ln2_b, h2_b);

    // 7) FFN
    k_gemm_ff1<<<2048, 256, 0, stream>>>(h2_b, w1_b, b1, ff1_b);
    k_gemm_ff2<<<512, 256, 0, stream>>>(ff1_b, w2_b, b2, x1, out);
}
